// CGTensorProductEquivariantModel_72645076844778
// MI455X (gfx1250) — matrix operand order repaired and checked by reading, not compiled, not run on hardware
//
#include <hip/hip_runtime.h>
#include <math.h>

// ---------------- problem constants (fixed by reference) ----------------
#define N_NODES   10000
#define N_EDGES   100000
#define NSC       48
#define NVC       10
#define IN_DIM    78            // NS + 3*NV
#define KDIM      144           // N_EDGE_FEAT
#define KPAD      160           // K padded to multiple of 32
#define KSTEPS    5             // KPAD / 32
#define WNUM      3364          // W_NUMEL
#define W_SS_END  2304
#define W_VS_END  2784
#define W_SV_END  3264
#define NCH1      9             // ceil(144/16)
#define NCH2      211           // ceil(3364/16)
#define EPW       32            // edges per wave (2 WMMA M-tiles)
#define NWAVES    4
#define EPB       (EPW*NWAVES)  // 128 edges per block
#define EPSV      1e-5f

#define A_SS 0.102062072616f    // 1/sqrt(96)
#define A_VS 0.223606797750f    // 1/sqrt(20)
#define A_SV 0.102062072616f
#define A_VV 0.223606797750f
#define INV_SQRT3 0.57735026919f

typedef __attribute__((ext_vector_type(16))) __bf16 v16bf;
typedef __attribute__((ext_vector_type(8)))  __bf16 bf16x8;
typedef __attribute__((ext_vector_type(8)))  float  v8f;
typedef __attribute__((ext_vector_type(4)))  int    i32x4;

// async global->LDS path (CDNA5), feature-guarded per compile pass
#if defined(__has_builtin)
#  if __has_builtin(__builtin_amdgcn_global_load_async_to_lds_b128) && \
      __has_builtin(__builtin_amdgcn_s_wait_asynccnt)
#    define USE_ASYNC 1
#  endif
#endif
#ifndef USE_ASYNC
#  define USE_ASYNC 0
#endif

// workspace layout (floats)
#define WS_SUMS   0
#define WS_CNT    780000
#define WS_PRM    790000        // 106 floats: scale[0..57], offset[58..105]
#define WS_W1     790112        // bf16 from here (16B aligned)
#define W1_ELEMS  (NCH1*KSTEPS*512)   // 23040 bf16
#define W2_ELEMS  (NCH2*KSTEPS*512)   // 540160 bf16

// ---------------- zero workspace ----------------
__global__ void zero_kernel(float* __restrict__ p, int n) {
    int i = blockIdx.x * blockDim.x + threadIdx.x;
    if (i < n) p[i] = 0.f;
}

// ------- swizzle fp32 weights [K,N] into WMMA-B bf16 fragment layout -------
//   col = chunk*16 + (lane&15),  k = s*32 + (lane<16 ? 0 : 16) + e
__global__ void swizzle_w(const float* __restrict__ src, __bf16* __restrict__ dst,
                          int K, int N, int chunks) {
    int total = chunks * KSTEPS * 512;
    for (int idx = blockIdx.x * blockDim.x + threadIdx.x; idx < total;
         idx += gridDim.x * blockDim.x) {
        int ch   = idx / (KSTEPS * 512);
        int rem  = idx % (KSTEPS * 512);
        int s    = rem / 512;
        int q    = rem % 512;
        int lane = q / 16;
        int el   = q % 16;
        int col  = ch * 16 + (lane & 15);
        int k    = s * 32 + ((lane < 16) ? 0 : 16) + el;
        float v  = (k < K && col < N) ? src[k * N + col] : 0.f;
        dst[idx] = (__bf16)v;
    }
}

// ---------------- fragment helpers ----------------
__device__ __forceinline__ v16bf load_frag16(const __bf16* p) {
    v16bf v;
    ((bf16x8*)&v)[0] = *(const bf16x8*)p;
    ((bf16x8*)&v)[1] = *(const bf16x8*)(p + 8);
    return v;
}

// A fragment straight from fp32 global edge_attr (row = one edge), bf16-convert
__device__ __forceinline__ v16bf load_a_global(const float* __restrict__ ea,
                                               int erow, int kb) {
    v16bf a;
    if (erow < N_EDGES) {
        const float* p = ea + (size_t)erow * KDIM;
#pragma unroll
        for (int u = 0; u < 8; ++u) {
            int k = kb + u;
            a[u] = (k < KDIM) ? (__bf16)p[k] : (__bf16)0.f;
        }
#pragma unroll
        for (int u = 0; u < 8; ++u) {
            int k = kb + 8 + u;
            a[u + 8] = (k < KDIM) ? (__bf16)p[k] : (__bf16)0.f;
        }
    } else {
#pragma unroll
        for (int u = 0; u < 16; ++u) a[u] = (__bf16)0.f;
    }
    return a;
}

#if USE_ASYNC
// wave0 copies one 5120B B-chunk global->LDS: 10 x b128 x 32 lanes
__device__ __forceinline__ void async_copy_chunk(const __bf16* __restrict__ src,
                                                 __bf16* dst, int lane) {
    const char* s = (const char*)src;
    char* d = (char*)dst;
#pragma unroll
    for (int j = 0; j < 10; ++j) {
        int off = (j * 32 + lane) * 16;
        __builtin_amdgcn_global_load_async_to_lds_b128(
            (__attribute__((address_space(1))) i32x4*)(s + off),
            (__attribute__((address_space(3))) i32x4*)(d + off), 0, 0);
    }
}
#endif

// ---------------- fused edge kernel: 4 waves x 32 edges per block ----------------
__global__ __launch_bounds__(128) void edge_kernel(
    const float* __restrict__ node_attr, const float* __restrict__ edge_attr,
    const float* __restrict__ edge_sh,
    const float* __restrict__ fc1_b, const float* __restrict__ fc2_b,
    const int*   __restrict__ eidx,
    const __bf16* __restrict__ w1f, const __bf16* __restrict__ w2f,
    float* __restrict__ sums, float* __restrict__ cnt)
{
    __shared__ __attribute__((aligned(16))) __bf16 hT[NWAVES][EPW * KPAD];     // 40KB
    __shared__ __attribute__((aligned(16))) __bf16 Bbuf[2][KSTEPS * 512];      // 10KB
    __shared__ float xs  [NWAVES][EPW][NSC];
    __shared__ float xv  [NWAVES][EPW][NVC][3];
    __shared__ float dots[NWAVES][EPW][NVC];
    __shared__ float s0a [NWAVES][EPW];
    __shared__ float s1a [NWAVES][EPW][3];
    __shared__ float outS[NWAVES][EPW][NSC];
    __shared__ float tSV [NWAVES][EPW][NVC];
    __shared__ float outV[NWAVES][EPW][NVC][3];
    __shared__ int   srcI[NWAVES][EPW];
    __shared__ int   dstI[NWAVES][EPW];

    const int tid  = threadIdx.x;
    const int wv   = tid >> 5;
    const int lane = tid & 31;
    const int we0  = blockIdx.x * EPB + wv * EPW;   // first edge of this wave

    // per-edge meta (one lane = one edge)
    {
        int e = we0 + lane;
        bool ok = e < N_EDGES;
        srcI[wv][lane]   = ok ? eidx[e] : -1;
        dstI[wv][lane]   = ok ? eidx[N_EDGES + e] : 0;
        s0a[wv][lane]    = ok ? edge_sh[e * 4 + 0] : 0.f;
        s1a[wv][lane][0] = ok ? edge_sh[e * 4 + 1] : 0.f;
        s1a[wv][lane][1] = ok ? edge_sh[e * 4 + 2] : 0.f;
        s1a[wv][lane][2] = ok ? edge_sh[e * 4 + 3] : 0.f;
    }
    for (int idx = lane; idx < EPW * KPAD; idx += 32) hT[wv][idx] = (__bf16)0.f;
    for (int idx = lane; idx < EPW * NSC;     idx += 32) (&outS[wv][0][0])[idx]    = 0.f;
    for (int idx = lane; idx < EPW * NVC;     idx += 32) (&tSV[wv][0][0])[idx]     = 0.f;
    for (int idx = lane; idx < EPW * NVC * 3; idx += 32) (&outV[wv][0][0][0])[idx] = 0.f;
    __syncthreads();

    // gather node features
    for (int idx = lane; idx < EPW * IN_DIM; idx += 32) {
        int e = idx / IN_DIM, f = idx % IN_DIM;
        float v = node_attr[(size_t)dstI[wv][e] * IN_DIM + f];
        if (f < NSC) xs[wv][e][f] = v;
        else { int q = f - NSC; xv[wv][e][q / 3][q % 3] = v; }
    }
    __syncthreads();

    // dot[e][i] = (xv . s1)/sqrt(3), pre-scaled by a_vs
    for (int idx = lane; idx < EPW * NVC; idx += 32) {
        int e = idx / NVC, i = idx % NVC;
        float d = (xv[wv][e][i][0] * s1a[wv][e][0] + xv[wv][e][i][1] * s1a[wv][e][1] +
                   xv[wv][e][i][2] * s1a[wv][e][2]) * INV_SQRT3;
        dots[wv][e][i] = d * A_VS;
    }
    __syncthreads();

    const int row = lane & 15;     // A row / C,D column index
    const int hi  = lane >> 4;     // lane half

    // ---- GEMM1: h = relu(edge_attr @ fc1_w + fc1_b), bf16 into LDS ----
    v16bf aF[2][KSTEPS];
#pragma unroll
    for (int t = 0; t < 2; ++t)
#pragma unroll
        for (int s = 0; s < KSTEPS; ++s)
            aF[t][s] = load_a_global(edge_attr, we0 + t * 16 + row, s * 32 + hi * 16);

    for (int ch = 0; ch < NCH1; ++ch) {
        v8f acc0 = {0.f,0.f,0.f,0.f,0.f,0.f,0.f,0.f};
        v8f acc1 = {0.f,0.f,0.f,0.f,0.f,0.f,0.f,0.f};
#pragma unroll
        for (int s = 0; s < KSTEPS; ++s) {
            v16bf b = load_frag16(w1f + ((size_t)(ch * KSTEPS + s) * 512 + lane * 16));
            acc0 = __builtin_amdgcn_wmma_f32_16x16x32_bf16(false, aF[0][s], false, b,
                                                           (short)0, acc0, false, false);
            acc1 = __builtin_amdgcn_wmma_f32_16x16x32_bf16(false, aF[1][s], false, b,
                                                           (short)0, acc1, false, false);
        }
        int col = ch * 16 + row;
        float b1 = fc1_b[col];
#pragma unroll
        for (int r = 0; r < 8; ++r) {
            float v0 = acc0[r] + b1; v0 = v0 > 0.f ? v0 : 0.f;
            float v1 = acc1[r] + b1; v1 = v1 > 0.f ? v1 : 0.f;
            hT[wv][(r + hi * 8) * KPAD + col]        = (__bf16)v0;
            hT[wv][(16 + r + hi * 8) * KPAD + col]   = (__bf16)v1;
        }
    }
    __syncthreads();

    // hoist all GEMM2 A fragments (whole per-wave h) into registers
    v16bf a2[2][KSTEPS];
#pragma unroll
    for (int t = 0; t < 2; ++t)
#pragma unroll
        for (int s = 0; s < KSTEPS; ++s)
            a2[t][s] = load_frag16(&hT[wv][(t * 16 + row) * KPAD + s * 32 + hi * 16]);

#if USE_ASYNC
    if (wv == 0) async_copy_chunk(w2f, &Bbuf[0][0], lane);   // prologue: chunk 0
#endif

    // ---- GEMM2 chunk loop: w-tile staged in LDS once per block ----
    for (int ch = 0; ch < NCH2; ++ch) {
        const int cur = ch & 1;
        __syncthreads();   // close all reads of Bbuf[cur^1] from previous iteration
#if USE_ASYNC
        if (wv == 0) {
            if (ch + 1 < NCH2) {
                async_copy_chunk(w2f + (size_t)(ch + 1) * KSTEPS * 512, &Bbuf[cur ^ 1][0], lane);
                __builtin_amdgcn_s_wait_asynccnt(10);   // chunk ch landed; ch+1 in flight
            } else {
                __builtin_amdgcn_s_wait_asynccnt(0);
            }
        }
#else
        {
            const unsigned long long* s8 =
                (const unsigned long long*)(w2f + (size_t)ch * KSTEPS * 512);
            unsigned long long* d8 = (unsigned long long*)&Bbuf[cur][0];
            for (int q = tid; q < (KSTEPS * 512) / 4; q += 128) d8[q] = s8[q];
        }
#endif
        __syncthreads();   // publish Bbuf[cur]

        v8f acc0 = {0.f,0.f,0.f,0.f,0.f,0.f,0.f,0.f};
        v8f acc1 = {0.f,0.f,0.f,0.f,0.f,0.f,0.f,0.f};
#pragma unroll
        for (int s = 0; s < KSTEPS; ++s) {
            v16bf b = load_frag16(&Bbuf[cur][s * 512 + lane * 16]);
            acc0 = __builtin_amdgcn_wmma_f32_16x16x32_bf16(false, a2[0][s], false, b,
                                                           (short)0, acc0, false, false);
            acc1 = __builtin_amdgcn_wmma_f32_16x16x32_bf16(false, a2[1][s], false, b,
                                                           (short)0, acc1, false, false);
        }

        // consume this 16-column w tile (m uniform per lane; 16 edge-rows held)
        int m = ch * 16 + row;
        if (m < WNUM) {
            float b2v = fc2_b[m];
            if (m < W_SS_END) {                       // w_ss[i,o]
                int i = m / NSC, o = m % NSC;
#pragma unroll
                for (int t = 0; t < 2; ++t)
#pragma unroll
                    for (int r = 0; r < 8; ++r) {
                        int e = t * 16 + r + hi * 8;
                        float wvv = (t ? acc1[r] : acc0[r]) + b2v;
                        atomicAdd(&outS[wv][e][o], A_SS * s0a[wv][e] * xs[wv][e][i] * wvv);
                    }
            } else if (m < W_VS_END) {                // w_vs[i,o]
                int tt = m - W_SS_END; int i = tt / NSC, o = tt % NSC;
#pragma unroll
                for (int t = 0; t < 2; ++t)
#pragma unroll
                    for (int r = 0; r < 8; ++r) {
                        int e = t * 16 + r + hi * 8;
                        float wvv = (t ? acc1[r] : acc0[r]) + b2v;
                        atomicAdd(&outS[wv][e][o], dots[wv][e][i] * wvv);
                    }
            } else if (m < W_SV_END) {                // w_sv[i,o]
                int tt = m - W_VS_END; int i = tt / NVC, o = tt % NVC;
#pragma unroll
                for (int t = 0; t < 2; ++t)
#pragma unroll
                    for (int r = 0; r < 8; ++r) {
                        int e = t * 16 + r + hi * 8;
                        float wvv = (t ? acc1[r] : acc0[r]) + b2v;
                        atomicAdd(&tSV[wv][e][o], A_SV * xs[wv][e][i] * wvv);
                    }
            } else {                                  // w_vv[i,o]
                int tt = m - W_SV_END; int i = tt / NVC, o = tt % NVC;
#pragma unroll
                for (int t = 0; t < 2; ++t)
#pragma unroll
                    for (int r = 0; r < 8; ++r) {
                        int e = t * 16 + r + hi * 8;
                        float c0 = A_VV * s0a[wv][e] * ((t ? acc1[r] : acc0[r]) + b2v);
                        atomicAdd(&outV[wv][e][o][0], c0 * xv[wv][e][i][0]);
                        atomicAdd(&outV[wv][e][o][1], c0 * xv[wv][e][i][1]);
                        atomicAdd(&outV[wv][e][o][2], c0 * xv[wv][e][i][2]);
                    }
            }
        }
    }
    __syncthreads();

    // ---- scatter tp into node segment sums (hardware FP atomics) ----
    for (int idx = lane; idx < EPW * IN_DIM; idx += 32) {
        int e = idx / IN_DIM, j = idx % IN_DIM;
        int sg = srcI[wv][e];
        if (sg >= 0) {
            float v;
            if (j < NSC) v = outS[wv][e][j];
            else { int q = j - NSC; int o = q / 3, c = q % 3;
                   v = outV[wv][e][o][c] + tSV[wv][e][o] * s1a[wv][e][c]; }
            unsafeAtomicAdd(&sums[(size_t)sg * IN_DIM + j], v);
        }
    }
    if (srcI[wv][lane] >= 0) unsafeAtomicAdd(&cnt[srcI[wv][lane]], 1.0f);
}

// ---------------- segment mean + residual ----------------
__global__ void node_combine(const float* __restrict__ sums, const float* __restrict__ cnt,
                             const float* __restrict__ node_attr, float* __restrict__ out) {
    int idx = blockIdx.x * blockDim.x + threadIdx.x;
    if (idx < N_NODES * IN_DIM) {
        int n = idx / IN_DIM;
        float c = cnt[n]; c = c < 1.f ? 1.f : c;
        out[idx] = sums[idx] / c + node_attr[idx];
    }
}

// ---------------- per-channel stats: block c reduces one channel ----------------
__global__ void stats_kernel(const float* __restrict__ out, const float* __restrict__ bnw,
                             const float* __restrict__ bnb, float* __restrict__ prm) {
    __shared__ float r1[256];
    __shared__ float r2[256];
    int c = blockIdx.x, t = threadIdx.x;
    float a1 = 0.f, a2 = 0.f;
    if (c < NSC) {
        for (int n = t; n < N_NODES; n += 256) {
            float v = out[n * IN_DIM + c]; a1 += v; a2 += v * v;
        }
    } else {
        int o = c - NSC;
        for (int n = t; n < N_NODES; n += 256) {
            const float* p = &out[n * IN_DIM + NSC + o * 3];
            a1 += p[0]*p[0] + p[1]*p[1] + p[2]*p[2];
        }
    }
    r1[t] = a1; r2[t] = a2; __syncthreads();
    for (int st = 128; st > 0; st >>= 1) {
        if (t < st) { r1[t] += r1[t + st]; r2[t] += r2[t + st]; }
        __syncthreads();
    }
    if (t == 0) {
        if (c < NSC) {
            float mean = r1[0] / (float)N_NODES;
            float var  = r2[0] / (float)N_NODES - mean * mean;
            float rstd = rsqrtf(var + EPSV);
            prm[c]      = rstd * bnw[c];                    // scale
            prm[58 + c] = bnb[c] - mean * rstd * bnw[c];    // offset
        } else {
            float fn = r1[0] / (3.f * (float)N_NODES);
            prm[c] = bnw[c] / sqrtf(fn + EPSV);             // vector scale
        }
    }
}

// ---------------- apply normalization ----------------
__global__ void norm_kernel(float* __restrict__ out, const float* __restrict__ prm) {
    int idx = blockIdx.x * blockDim.x + threadIdx.x;
    if (idx < N_NODES * IN_DIM) {
        int j = idx % IN_DIM;
        if (j < NSC) out[idx] = out[idx] * prm[j] + prm[58 + j];
        else { int o = (j - NSC) / 3; out[idx] *= prm[NSC + o]; }
    }
}

// ---------------- launch ----------------
extern "C" void kernel_launch(void* const* d_in, const int* in_sizes, int n_in,
                              void* d_out, int out_size, void* d_ws, size_t ws_size,
                              hipStream_t stream) {
    const float* node_attr = (const float*)d_in[0];
    const float* edge_attr = (const float*)d_in[1];
    const float* edge_sh   = (const float*)d_in[2];
    const float* fc1_w     = (const float*)d_in[3];
    const float* fc1_b     = (const float*)d_in[4];
    const float* fc2_w     = (const float*)d_in[5];
    const float* fc2_b     = (const float*)d_in[6];
    const float* bnw       = (const float*)d_in[7];
    const float* bnb       = (const float*)d_in[8];
    const int*   eidx      = (const int*)d_in[9];
    float* out = (float*)d_out;

    float*  wsf  = (float*)d_ws;
    float*  sums = wsf + WS_SUMS;
    float*  cntp = wsf + WS_CNT;
    float*  prm  = wsf + WS_PRM;
    __bf16* w1f  = (__bf16*)(wsf + WS_W1);
    __bf16* w2f  = w1f + W1_ELEMS;

    const int nzero = WS_PRM + 106;   // sums + cnt + params
    zero_kernel<<<(nzero + 255) / 256, 256, 0, stream>>>(wsf, nzero);

    swizzle_w<<<(W1_ELEMS + 255) / 256, 256, 0, stream>>>(fc1_w, w1f, KDIM, KDIM, NCH1);
    swizzle_w<<<(W2_ELEMS + 255) / 256, 256, 0, stream>>>(fc2_w, w2f, KDIM, WNUM, NCH2);

    edge_kernel<<<(N_EDGES + EPB - 1) / EPB, 128, 0, stream>>>(
        node_attr, edge_attr, edge_sh, fc1_b, fc2_b, eidx, w1f, w2f, sums, cntp);

    node_combine<<<(N_NODES * IN_DIM + 255) / 256, 256, 0, stream>>>(sums, cntp, node_attr, out);
    stats_kernel<<<NSC + NVC, 256, 0, stream>>>(out, bnw, bnb, prm);
    norm_kernel<<<(N_NODES * IN_DIM + 255) / 256, 256, 0, stream>>>(out, prm);
}
